// Quantize_57638461112474
// MI455X (gfx1250) — matrix operand order repaired and checked by reading, not compiled, not run on hardware
//
#include <hip/hip_runtime.h>
#include <hip/hip_bf16.h>

// ---- sizes (fixed by the reference) ----
#define CODE_SIZE 1024
#define CDIM      256
#define NPIX      65536            // B*H*W = 64*32*32
#define QELEMS    4194304          // B*C*H*W = quantized element count
#define PIX_WG    32               // pixels per workgroup
#define A_STRIDE  264              // padded LDS row stride (bf16 elems): 264*2B=528B -> bank offset 4/row

typedef __bf16          v16bf __attribute__((ext_vector_type(16)));
typedef float           v8f   __attribute__((ext_vector_type(8)));
typedef unsigned short  us8   __attribute__((ext_vector_type(8)));
typedef unsigned short  us16  __attribute__((ext_vector_type(16)));

__device__ __forceinline__ unsigned short f32_to_bf16_rne(float f) {
    unsigned int u = __float_as_uint(f);
    unsigned int r = u + 0x7FFFu + ((u >> 16) & 1u);
    return (unsigned short)(r >> 16);
}
__device__ __forceinline__ float bf16us_to_f32(unsigned short h) {
    return __uint_as_float(((unsigned int)h) << 16);
}
__device__ __forceinline__ v16bf cat16(us8 a, us8 b) {
    us16 u = __builtin_shufflevector(a, b, 0,1,2,3,4,5,6,7,8,9,10,11,12,13,14,15);
    return __builtin_bit_cast(v16bf, u);
}

// ---------------------------------------------------------------------------
// Prep: split weight (1024x256 f32) into bf16 hi/lo and compute ||w||^2.
// ---------------------------------------------------------------------------
__global__ __launch_bounds__(256)
void vq_prep_kernel(const float* __restrict__ weight,
                    unsigned short* __restrict__ Whi,
                    unsigned short* __restrict__ Wlo,
                    float* __restrict__ wnorm) {
    __shared__ float red[256];
    const int j = blockIdx.x;      // code id
    const int t = threadIdx.x;     // feature id
    float v = weight[j * CDIM + t];
    unsigned short hi = f32_to_bf16_rne(v);
    float rem = v - bf16us_to_f32(hi);
    Whi[j * CDIM + t] = hi;
    Wlo[j * CDIM + t] = f32_to_bf16_rne(rem);
    red[t] = v * v;
    __syncthreads();
    for (int s = 128; s > 0; s >>= 1) {
        if (t < s) red[t] += red[t + s];
        __syncthreads();
    }
    if (t == 0) wnorm[j] = red[0];
}

// ---------------------------------------------------------------------------
// Main: 32 pixels/WG x 1024 codes. 8 waves, each wave owns 128 codes.
// Split-bf16 WMMA GEMM -> score -> argmin -> gather -> write q/ste/idx.
// ---------------------------------------------------------------------------
__global__ __launch_bounds__(256)
void vq_main_kernel(const float* __restrict__ x,
                    const float* __restrict__ weight,
                    const unsigned short* __restrict__ Whi,
                    const unsigned short* __restrict__ Wlo,
                    const float* __restrict__ wnorm,
                    float* __restrict__ out) {
    __shared__ unsigned short sAhi[PIX_WG * A_STRIDE];
    __shared__ unsigned short sAlo[PIX_WG * A_STRIDE];
    __shared__ float sRedV[8 * PIX_WG];
    __shared__ int   sRedI[8 * PIX_WG];
    __shared__ int   sIdx[PIX_WG];

    const int t    = threadIdx.x;
    const int wv   = t >> 5;       // wave id 0..7
    const int lane = t & 31;
    const int tile = blockIdx.x;   // 2048 tiles of 32 pixels
    const int b     = tile >> 5;           // image (1024 pixels each, 32|1024)
    const int pbase = (tile & 31) * PIX_WG;
    const float* xb = x + (size_t)b * (CDIM * 1024) + pbase;

    // ---- Stage A tile (32 pixels x 256 features) as bf16 hi/lo in LDS ----
    {
        const int pix = t & 31;     // lane-consecutive pixels -> coalesced
        const int cg  = t >> 5;
        for (int i = 0; i < 32; ++i) {
            int c = cg * 32 + i;
            float v = xb[(size_t)c * 1024 + pix];
            unsigned short hi = f32_to_bf16_rne(v);
            float rem = v - bf16us_to_f32(hi);
            sAhi[pix * A_STRIDE + c] = hi;
            sAlo[pix * A_STRIDE + c] = f32_to_bf16_rne(rem);
        }
    }
    __syncthreads();

    const int m    = lane & 15;    // row/col within tile
    const int half = lane >> 4;

    float bestV[2][8];
    int   bestI[2][8];
#pragma unroll
    for (int a = 0; a < 2; ++a)
#pragma unroll
        for (int r = 0; r < 8; ++r) { bestV[a][r] = 3.0e38f; bestI[a][r] = 0; }

    for (int cb = 0; cb < 2; ++cb) {
        const int codebase = wv * 128 + cb * 64;

        v8f acc[2][4];
#pragma unroll
        for (int a = 0; a < 2; ++a)
#pragma unroll
            for (int bt = 0; bt < 4; ++bt)
#pragma unroll
                for (int r = 0; r < 8; ++r) acc[a][bt][r] = 0.0f;

        for (int kk = 0; kk < 8; ++kk) {   // K = 256 in chunks of 32
            const int kb = kk * 32;
            // A operands: lane m holds row, elems 0-15 -> K=half*16+e
            v16bf Ahi[2], Alo[2];
#pragma unroll
            for (int a = 0; a < 2; ++a) {
                int row = a * 16 + m;
                int o0 = row * A_STRIDE + kb + half * 16;
                int o1 = o0 + 8;
                Ahi[a] = cat16(*(const us8*)(sAhi + o0), *(const us8*)(sAhi + o1));
                Alo[a] = cat16(*(const us8*)(sAlo + o0), *(const us8*)(sAlo + o1));
            }
#pragma unroll
            for (int bt = 0; bt < 4; ++bt) {
                // B operand: lane m holds code column, elems 0-15 -> K=half*16+e (contiguous in W row)
                int code = codebase + bt * 16 + m;
                const unsigned short* ph = Whi + (size_t)code * CDIM + kb + half * 16;
                const unsigned short* pl = Wlo + (size_t)code * CDIM + kb + half * 16;
                v16bf Bhi = cat16(*(const us8*)ph, *(const us8*)(ph + 8));
                v16bf Blo = cat16(*(const us8*)pl, *(const us8*)(pl + 8));
#pragma unroll
                for (int a = 0; a < 2; ++a) {
                    // split product: hi*hi + hi*lo + lo*hi  (lo*lo negligible)
                    acc[a][bt] = __builtin_amdgcn_wmma_f32_16x16x32_bf16(
                        false, Ahi[a], false, Bhi, (short)0, acc[a][bt], false, false);
                    acc[a][bt] = __builtin_amdgcn_wmma_f32_16x16x32_bf16(
                        false, Ahi[a], false, Blo, (short)0, acc[a][bt], false, false);
                    acc[a][bt] = __builtin_amdgcn_wmma_f32_16x16x32_bf16(
                        false, Alo[a], false, Bhi, (short)0, acc[a][bt], false, false);
                }
            }
        }

        // fold scores: d = ||w||^2 - 2 x.w  (||x||^2 constant per row, dropped)
#pragma unroll
        for (int bt = 0; bt < 4; ++bt) {
            int code = codebase + bt * 16 + m;
            float wn = wnorm[code];
#pragma unroll
            for (int a = 0; a < 2; ++a)
#pragma unroll
                for (int r = 0; r < 8; ++r) {
                    float s = wn - 2.0f * acc[a][bt][r];
                    if (s < bestV[a][r]) { bestV[a][r] = s; bestI[a][r] = code; }
                }
        }
    }

    // ---- butterfly min+argmin across the 16 lanes of each half-group ----
#pragma unroll
    for (int a = 0; a < 2; ++a)
#pragma unroll
        for (int r = 0; r < 8; ++r) {
            float v = bestV[a][r];
            int   i = bestI[a][r];
#pragma unroll
            for (int mask = 1; mask <= 8; mask <<= 1) {
                float ov = __shfl_xor(v, mask, 32);
                int   oi = __shfl_xor(i, mask, 32);
                if (ov < v || (ov == v && oi < i)) { v = ov; i = oi; }
            }
            bestV[a][r] = v; bestI[a][r] = i;
        }

    // per-wave result: lane 0 -> rows a*16+r, lane 16 -> rows a*16+8+r
    if (m == 0) {
#pragma unroll
        for (int a = 0; a < 2; ++a)
#pragma unroll
            for (int r = 0; r < 8; ++r) {
                int pixel = a * 16 + half * 8 + r;
                sRedV[wv * PIX_WG + pixel] = bestV[a][r];
                sRedI[wv * PIX_WG + pixel] = bestI[a][r];
            }
    }
    __syncthreads();

    // ---- cross-wave reduce + write idx (as float, 3rd output) ----
    if (t < PIX_WG) {
        float v = sRedV[t];
        int   i = sRedI[t];
        for (int w2 = 1; w2 < 8; ++w2) {
            float ov = sRedV[w2 * PIX_WG + t];
            int   oi = sRedI[w2 * PIX_WG + t];
            if (ov < v || (ov == v && oi < i)) { v = ov; i = oi; }
        }
        sIdx[t] = i;
        out[(size_t)2 * QELEMS + (size_t)tile * PIX_WG + t] = (float)i;
    }
    __syncthreads();

    // ---- gather weight[idx] and write quantized + ste (B,C,H,W layout) ----
    {
        const int pix = t & 31;
        const int cg  = t >> 5;
        const int idx = sIdx[pix];
        const float* wrow = weight + (size_t)idx * CDIM;
        const size_t obase = (size_t)b * (CDIM * 1024) + pbase + pix;
        for (int i = 0; i < 32; ++i) {
            int c = cg * 32 + i;
            float v = wrow[c];
            out[obase + (size_t)c * 1024] = v;                    // quantized
            out[(size_t)QELEMS + obase + (size_t)c * 1024] = v;   // ste == quantized
        }
    }
}

// ---------------------------------------------------------------------------
extern "C" void kernel_launch(void* const* d_in, const int* in_sizes, int n_in,
                              void* d_out, int out_size, void* d_ws, size_t ws_size,
                              hipStream_t stream) {
    const float* x      = (const float*)d_in[0];   // (64,256,32,32)
    const float* weight = (const float*)d_in[1];   // (1024,256)
    float* out = (float*)d_out;                    // quantized | ste | idx(float)

    // workspace layout: Whi (512KB) | Wlo (512KB) | wnorm (4KB)  -> ~1.03 MB
    unsigned short* Whi = (unsigned short*)d_ws;
    unsigned short* Wlo = (unsigned short*)((char*)d_ws + (size_t)CODE_SIZE * CDIM * 2);
    float* wnorm        = (float*)((char*)d_ws + (size_t)CODE_SIZE * CDIM * 4);

    vq_prep_kernel<<<CODE_SIZE, 256, 0, stream>>>(weight, Whi, Wlo, wnorm);
    vq_main_kernel<<<NPIX / PIX_WG, 256, 0, stream>>>(x, weight, Whi, Wlo, wnorm, out);
}
